// LMM_12094627906359
// MI455X (gfx1250) — matrix operand order repaired and checked by reading, not compiled, not run on hardware
//
#include <hip/hip_runtime.h>
#include <stdint.h>

typedef __attribute__((ext_vector_type(16))) _Float16 v16h;
typedef __attribute__((ext_vector_type(8)))  float    v8f;
typedef _Float16 half_t;

#define BB      2
#define NN      20000
#define CC      256
#define KK      5000
#define KPAD    5008          // K padded to multiple of 16 for WMMA N-tiles
#define NTILES  (KPAD / 16)   // 313
#define NLAYERS 6
#define NTW     4             // n-tiles per wave (16x64 output strip)

// ---------------------------------------------------------------------------
// 1) Pack fp32 weights -> fp16 (row-major [o][i], 6 layers concatenated)
// ---------------------------------------------------------------------------
__global__ void k_pack_w(const float* __restrict__ W, half_t* __restrict__ Wh, int n) {
    int i = blockIdx.x * 256 + threadIdx.x;
    if (i < n) Wh[i] = (half_t)W[i];
}

// ---------------------------------------------------------------------------
// 2) Exact top-k via stable rank counting (matches jax.lax.top_k ordering):
//    rank(i) = #{j : s_j > s_i} + #{j < i : s_j == s_i}
// ---------------------------------------------------------------------------
__global__ void k_rank(const float* __restrict__ score, int* __restrict__ rank) {
    __shared__ float tile[256];
    const int b   = blockIdx.y;
    const int i   = blockIdx.x * 256 + threadIdx.x;
    const float* s = score + b * NN;
    const bool valid = (i < NN);
    const float si = valid ? s[i] : 0.0f;
    int cnt = 0;
    for (int base = 0; base < NN; base += 256) {
        int j = base + threadIdx.x;
        tile[threadIdx.x] = (j < NN) ? s[j] : -3.4e38f;
        __syncthreads();
        if (valid) {
            #pragma unroll 4
            for (int jj = 0; jj < 256; ++jj) {
                float sj = tile[jj];
                int   j2 = base + jj;
                cnt += (int)((sj > si) || (sj == si && j2 < i));
            }
        }
        __syncthreads();
    }
    if (valid) rank[b * NN + i] = cnt;
}

// ---------------------------------------------------------------------------
// 3) Scatter kept entries: source index, boxes, and `keep` output (as float)
// ---------------------------------------------------------------------------
__global__ void k_scatter(const float* __restrict__ box, const int* __restrict__ rank,
                          int* __restrict__ srcIdx, float* __restrict__ boxK,
                          float* __restrict__ keep_out) {
    const int b = blockIdx.y;
    const int i = blockIdx.x * 256 + threadIdx.x;
    if (i >= NN) return;
    const int r = rank[b * NN + i];
    if (r < KK) {
        srcIdx[b * KK + r] = i;
        #pragma unroll
        for (int c = 0; c < 4; ++c)
            boxK[(b * KK + r) * 4 + c] = box[(b * 4 + c) * NN + i];
        keep_out[b * KK + r] = (float)i;
    }
}

// ---------------------------------------------------------------------------
// 4) Gather features into f16 activation buffer, layout [b][k][c], zero pad
// ---------------------------------------------------------------------------
__global__ void k_gather_feat(const float* __restrict__ feat, const int* __restrict__ srcIdx,
                              half_t* __restrict__ x) {
    const int bk = blockIdx.x;
    const int b  = bk / KPAD;
    const int k  = bk % KPAD;
    const int c  = threadIdx.x;
    float v = 0.0f;
    if (k < KK) {
        int i = srcIdx[b * KK + k];
        v = feat[(b * CC + c) * NN + i];
    }
    x[(size_t)(b * KPAD + k) * CC + c] = (half_t)v;
}

// ---------------------------------------------------------------------------
// 5) NMS-style suppression: m_j = max_{i<j} IoU(box_i, box_j)
//    (scores sorted desc => argmax of adjc reduces to min-index with iou>=thr)
// ---------------------------------------------------------------------------
__global__ void k_suppress(const float* __restrict__ boxK, float* __restrict__ mj) {
    __shared__ float4 tb[256];
    const int b = blockIdx.y;
    const int j = blockIdx.x * 256 + threadIdx.x;
    const float4* bx = (const float4*)boxK;
    float4 bj = make_float4(0.f, 0.f, 1.f, 1.f);
    float areaj = 1.0f;
    const bool valid = (j < KK);
    if (valid) {
        bj = bx[b * KK + j];
        areaj = (bj.z - bj.x) * (bj.w - bj.y);
    }
    float m = 0.0f;
    const int last = (int)blockIdx.x * 256;
    for (int base = 0; base <= last; base += 256) {
        int idx = base + threadIdx.x;
        tb[threadIdx.x] = (idx < KK) ? bx[b * KK + idx] : make_float4(0.f, 0.f, 1.f, 1.f);
        __syncthreads();
        int lim = j - base; if (lim > 256) lim = 256;
        for (int t = 0; t < lim; ++t) {
            float4 bi = tb[t];
            float areai = (bi.z - bi.x) * (bi.w - bi.y);
            float lx = fmaxf(bi.x, bj.x), ly = fmaxf(bi.y, bj.y);
            float rx = fminf(bi.z, bj.z), ry = fminf(bi.w, bj.w);
            float w = fmaxf(rx - lx, 0.0f), h = fmaxf(ry - ly, 0.0f);
            float inter = w * h;
            float iou = inter / (areai + areaj - inter);
            m = fmaxf(m, iou);
        }
        __syncthreads();
    }
    if (valid) mj[b * KK + j] = m;
}

// ---------------------------------------------------------------------------
// 6) WMMA GEMM layer: y[o,k] = leaky(sum_i W[o,i] * x[i,k] + bias[o], 0.2)
//    One wave computes a 16(M) x 64(N) strip: 4 accumulators share one A
//    fragment per k-step -> 4 WMMAs per A-load, 4x less W traffic, and 4
//    independent WMMAs for the scheduler to pipeline (no D->A RAW chains).
//    A/B k order: contiguous k = 16*hi + s in both; D layout per CDNA5 ISA 7.12.2 (wave32).
// ---------------------------------------------------------------------------
__global__ __launch_bounds__(128)
void k_gemm(const half_t* __restrict__ xin, half_t* __restrict__ xout,
            const half_t* __restrict__ Wl, const float* __restrict__ bias) {
    const int wave  = threadIdx.x >> 5;
    const int lane  = threadIdx.x & 31;
    const int hi    = lane >> 4;      // half-wave: 0 or 1
    const int lm    = lane & 15;
    const int nt0   = (blockIdx.x * 4 + wave) * NTW;  // first n-tile of strip
    const int mtile = blockIdx.y;
    const int b     = blockIdx.z;

    const int mrow = mtile * 16 + lm;                 // A-matrix row (out chan)
    const uint32_t* W32 = (const uint32_t*)Wl;        // 2 halves per dword

    // Per-column base pointers; KPAD = NTILES*16 exactly, so out-of-range is
    // tile-uniform: clamp the load pointer, skip the store.
    const uint4* xq[NTW];
    int ncol[NTW];
    #pragma unroll
    for (int t = 0; t < NTW; ++t) {
        const int nt = nt0 + t;
        ncol[t] = nt * 16 + lm;
        const int nld = (nt < NTILES) ? ncol[t] : lm; // clamped, loads stay in-bounds
        xq[t] = (const uint4*)(xin + (size_t)(b * KPAD + nld) * CC);
    }

    __builtin_prefetch(Wl + mrow * CC, 0, 0);         // global_prefetch_b8

    v8f acc[NTW] = {};
    #pragma unroll
    for (int s = 0; s < 8; ++s) {                     // 8 k-steps of 32
        const int ks = s * 32;
        // A fragment: lane holds row `mrow`; VGPR r -> K pair hi*16 + 2r (contiguous, matches B)
        union { v16h v; uint32_t u[8]; } A;
        #pragma unroll
        for (int r = 0; r < 8; ++r) {
            int kb = 2 * r + hi * 16;
            A.u[r] = W32[(mrow * CC + ks + kb) >> 1];
        }
        // B fragments: lane holds column, contiguous 16 K-values at hi*16
        const int c0 = (ks + hi * 16) / 8;            // uint4 chunk index
        union { v16h v; uint4 q[2]; } Bf[NTW];
        #pragma unroll
        for (int t = 0; t < NTW; ++t) {
            Bf[t].q[0] = xq[t][c0];
            Bf[t].q[1] = xq[t][c0 + 1];
        }
        #pragma unroll
        for (int t = 0; t < NTW; ++t)
            acc[t] = __builtin_amdgcn_wmma_f32_16x16x32_f16(
                         false, A.v, false, Bf[t].v, (short)0, acc[t], false, false);
    }

    // D layout: VGPR v holds (m = mtile*16 + hi*8 + v, n = ntile*16 + lm)
    const int obase = mtile * 16 + hi * 8;
    float bv[8];
    #pragma unroll
    for (int v = 0; v < 8; ++v) bv[v] = bias[obase + v];
    #pragma unroll
    for (int t = 0; t < NTW; ++t) {
        if (nt0 + t >= NTILES) continue;              // wave-uniform guard
        union { uint4 q; half_t h[8]; } out;
        #pragma unroll
        for (int v = 0; v < 8; ++v) {
            float y = acc[t][v] + bv[v];
            y = (y > 0.0f) ? y : 0.2f * y;            // leaky_relu(0.2)
            out.h[v] = (half_t)y;
        }
        *(uint4*)(xout + (size_t)(b * KPAD + ncol[t]) * CC + obase) = out.q;
    }
}

// ---------------------------------------------------------------------------
// 7) Final head: 3x256 linear + softmax over 3, combine with NMS masks
// ---------------------------------------------------------------------------
__global__ void k_final(const half_t* __restrict__ x, const float* __restrict__ Wf,
                        const float* __restrict__ bf, const float* __restrict__ mj,
                        float* __restrict__ out_loc) {
    const int b = blockIdx.y;
    const int k = blockIdx.x * 256 + threadIdx.x;
    if (k >= KK) return;
    const half_t* xc = x + (size_t)(b * KPAD + k) * CC;
    float d0 = bf[0], d1 = bf[1], d2 = bf[2];
    for (int c = 0; c < CC; ++c) {
        float v = (float)xc[c];
        d0 += Wf[0 * CC + c] * v;
        d1 += Wf[1 * CC + c] * v;
        d2 += Wf[2 * CC + c] * v;
    }
    float mx = fmaxf(d0, fmaxf(d1, d2));
    float e0 = __expf(d0 - mx), e1 = __expf(d1 - mx), e2 = __expf(d2 - mx);
    float inv = 1.0f / (e0 + e1 + e2);
    float m = mj[b * KK + k];
    float loc = 0.0f;
    if (m < 0.4f) loc += e0 * inv;   // mask_t = (max_{i<j} iou < thr)
    if (m < 0.6f) loc += e1 * inv;
    if (m < 0.8f) loc += e2 * inv;
    out_loc[b * KK + k] = loc;
}

// ---------------------------------------------------------------------------
extern "C" void kernel_launch(void* const* d_in, const int* in_sizes, int n_in,
                              void* d_out, int out_size, void* d_ws, size_t ws_size,
                              hipStream_t stream) {
    const float* box   = (const float*)d_in[0];  // [B,4,N]
    const float* score = (const float*)d_in[1];  // [B,1,N]
    const float* feat  = (const float*)d_in[2];  // [B,C,N]
    const float* W     = (const float*)d_in[3];  // [6,C,C]
    const float* bias  = (const float*)d_in[4];  // [6,C]
    const float* Wf    = (const float*)d_in[5];  // [3,C]
    const float* bf    = (const float*)d_in[6];  // [3]
    float* out = (float*)d_out;                  // [B*K] loc_max ++ [B*K] keep

    // Workspace carve-up (256B aligned)
    char*  ws  = (char*)d_ws;
    size_t off = 0;
    auto carve = [&](size_t bytes) -> char* {
        char* p = ws + off;
        off = (off + bytes + 255) & ~(size_t)255;
        return p;
    };
    half_t* Wh     = (half_t*)carve((size_t)NLAYERS * CC * CC * sizeof(half_t));
    half_t* xA     = (half_t*)carve((size_t)BB * KPAD * CC * sizeof(half_t));
    half_t* xB     = (half_t*)carve((size_t)BB * KPAD * CC * sizeof(half_t));
    int*    rank   = (int*)   carve((size_t)BB * NN * sizeof(int));
    int*    srcIdx = (int*)   carve((size_t)BB * KK * sizeof(int));
    float*  boxK   = (float*) carve((size_t)BB * KK * 4 * sizeof(float));
    float*  mj     = (float*) carve((size_t)BB * KK * sizeof(float));
    (void)in_sizes; (void)n_in; (void)out_size; (void)ws_size;

    // 1) pack weights to f16
    {
        int n = NLAYERS * CC * CC;
        k_pack_w<<<dim3((n + 255) / 256), dim3(256), 0, stream>>>(W, Wh, n);
    }
    // 2) exact stable top-k ranks
    k_rank<<<dim3((NN + 255) / 256, BB), dim3(256), 0, stream>>>(score, rank);
    // 3) scatter kept boxes / indices / keep output
    k_scatter<<<dim3((NN + 255) / 256, BB), dim3(256), 0, stream>>>(
        box, rank, srcIdx, boxK, out + (size_t)BB * KK);
    // 4) gather features (f16, [b][k][c], zero-padded columns)
    k_gather_feat<<<dim3(BB * KPAD), dim3(256), 0, stream>>>(feat, srcIdx, xA);
    // 5) suppression running-max IoU
    k_suppress<<<dim3((KK + 255) / 256, BB), dim3(256), 0, stream>>>(boxK, mj);
    // 6) six WMMA GEMM layers, ping-pong xA <-> xB (ends back in xA)
    //    grid.x: each block = 4 waves x 4 n-tiles = 16 n-tiles -> ceil(313/16)=20
    half_t* bufs[2] = { xA, xB };
    const int ngroups = (NTILES + 4 * NTW - 1) / (4 * NTW);
    for (int l = 0; l < NLAYERS; ++l) {
        k_gemm<<<dim3(ngroups, CC / 16, BB), dim3(128), 0, stream>>>(
            bufs[l & 1], bufs[(l + 1) & 1], Wh + (size_t)l * CC * CC, bias + l * CC);
    }
    // 7) final head + softmax + mask combine
    k_final<<<dim3((KK + 255) / 256, BB), dim3(256), 0, stream>>>(
        bufs[NLAYERS & 1], Wf, bf, mj, out);
}
